// UNet_7834020348505
// MI455X (gfx1250) — matrix operand order repaired and checked by reading, not compiled, not run on hardware
//
#include <hip/hip_runtime.h>
#include <hip/hip_bf16.h>
#include <stdint.h>

typedef __attribute__((ext_vector_type(16))) __bf16 v16bf;
typedef __attribute__((ext_vector_type(8)))  float  v8f;

// ---------------- problem dims ----------------
// x  : (1,1,12,48,96,96) f32
// h  : (10,46,94,94,16)  bf16 channels-last intermediate (in d_ws)
// out: (1,16,8,44,92,92) f32
#define XL 12
#define XD 48
#define XH 96
#define XW 96
#define HL 10
#define HD 46
#define HH 94
#define HW 94
#define NP1 (HL*HD*HH*HW)              // 4,064,560 layer-1 positions
#define OL 8
#define OD 44
#define OH 92
#define OW 92
#define WT 6                           // W tiles of 16 covering 92
#define NSTEP 41                       // ceil(81 taps / 2)
#define OUT_NSTRIDE (OL*OD*OH*OW)      // 2,979,328 (per-out-channel stride)
#define HBYTES ((size_t)NP1 * 16 * 2)  // 130,065,920 bytes of bf16 h

__device__ __forceinline__ unsigned f2bf1(float f) {
  unsigned u = __builtin_bit_cast(unsigned, f);
  return (u + 0x7FFFu + ((u >> 16) & 1u)) >> 16;  // round-to-nearest-even
}

// ---------------- w2 -> B-fragment pack ----------------
// B fragment layout for v_wmma_f32_16x16x32_bf16 (wave32):
//   value B[k][n] lives in lane = n + 16*((k%16)/8), element e = 8*(k/16) + k%8.
// Here k = 16*(tap - t0) + c  (2 taps per step), so:
//   frag[s][lane][e] = bf16( w2[o = lane%16][i = 8*(lane/16) + e%8][tap = 2s + e/8] )
// tap 81 (pad of the odd 81-tap count) is zero.
__global__ __launch_bounds__(256) void pack_w2_kernel(
    const float* __restrict__ w2, unsigned short* __restrict__ frag) {
  int idx = blockIdx.x * 256 + threadIdx.x;     // over 41*32*16 = 20,992
  if (idx >= NSTEP * 32 * 16) return;
  int e    = idx & 15;
  int lane = (idx >> 4) & 31;
  int s    = idx >> 9;
  int n    = lane & 15;
  int hl   = lane >> 4;
  int t    = 2 * s + (e >> 3);
  float v = (t <= 80) ? w2[(size_t)(n * 16 + 8 * hl + (e & 7)) * 81 + t] : 0.0f;
  frag[idx] = (unsigned short)f2bf1(v);
}

// ---------------- layer 1: direct conv (Cin=1), f32 -> bf16 channels-last ----------------
__global__ __launch_bounds__(256) void conv1_kernel(
    const float* __restrict__ x, const float* __restrict__ w1,
    const float* __restrict__ b1, unsigned short* __restrict__ h) {
  __shared__ float wsm[81 * 16];   // [tap][outC]
  __shared__ float bsum[16];
  for (int i = threadIdx.x; i < 81 * 16; i += 256) {
    int t = i >> 4, c = i & 15;
    wsm[i] = w1[c * 81 + t];       // w1 flat: (o, i=0, tl, td, th, tw) = o*81 + t
  }
  if (threadIdx.x < 16) {
    int c = threadIdx.x;
    bsum[c] = b1[c] + b1[16 + c] + b1[32 + c];   // b1.sum(0)
  }
  __syncthreads();

  int p = blockIdx.x * 256 + threadIdx.x;
  if (p >= NP1) return;
  int w  = p % HW;
  int r  = p / HW;
  int hh = r % HH; r /= HH;
  int d  = r % HD;
  int l  = r / HD;

  const float* xb = x + (size_t)l * (XD * XH * XW) + (size_t)d * (XH * XW) + hh * XW + w;
  float acc[16];
#pragma unroll
  for (int c = 0; c < 16; ++c) acc[c] = bsum[c];

  int t = 0;
  for (int tl = 0; tl < 3; ++tl)
    for (int td = 0; td < 3; ++td)
      for (int th = 0; th < 3; ++th) {
        const float* xr = xb + tl * (XD * XH * XW) + td * (XH * XW) + th * XW;
#pragma unroll
        for (int tw = 0; tw < 3; ++tw) {
          float xv = xr[tw];
          const float* wr = &wsm[(t + tw) * 16];
#pragma unroll
          for (int c = 0; c < 16; ++c) acc[c] = fmaf(xv, wr[c], acc[c]);
        }
        t += 3;
      }

  unsigned pk[8];
#pragma unroll
  for (int j = 0; j < 8; ++j) {
    float a0 = fmaxf(acc[2 * j + 0], 0.0f);
    float a1 = fmaxf(acc[2 * j + 1], 0.0f);
    pk[j] = f2bf1(a0) | (f2bf1(a1) << 16);
  }
  uint4* hp = (uint4*)(h + (size_t)p * 16);   // 32B, 16B-aligned
  hp[0] = make_uint4(pk[0], pk[1], pk[2], pk[3]);
  hp[1] = make_uint4(pk[4], pk[5], pk[6], pk[7]);
}

// ---------------- layer 2: implicit-GEMM WMMA conv ----------------
// byte offset inside h for tap t relative to the output position
__device__ __forceinline__ int tapOffB(int t) {
  int tl = t / 27; int r = t - tl * 27;
  int td = r / 9;  r -= td * 9;
  int th = r / 3;  int tw = r - th * 3;
  return (((tl * HD + td) * HH + th) * HW + tw) * 32;   // 32 B = 16 bf16 channels
}

__global__ __launch_bounds__(128) void conv2_kernel(
    const unsigned short* __restrict__ h, const unsigned short* __restrict__ frag,
    const float* __restrict__ b2, float* __restrict__ out) {
  __shared__ float xpose[4][16 * 17];

  const int lane = threadIdx.x & 31;
  const int wv   = threadIdx.x >> 5;
  int wid = blockIdx.x * 4 + wv;                 // one wave = one 16x16 output tile
  int wt = wid % WT; int r = wid / WT;
  int hh = r % OH;   r /= OH;
  int d  = r % OD;
  int l  = r / OD;                                // exact: grid covers 8*44*92*6 waves

  const int n  = lane & 15;                       // also the A-row m this lane feeds
  const int hl = lane >> 4;                       // lane half selects K sub-range

  // A base: position (l, d, hh, w0 + m), channel 8*hl, tap added per step
  const char* hbase = (const char*)h
      + ((size_t)(((l * HD + d) * HH + hh) * HW + wt * 16 + n) * 32)
      + hl * 16;
  const char* fbase = (const char*)frag + (size_t)lane * 32;

  v8f acc = {0.f, 0.f, 0.f, 0.f, 0.f, 0.f, 0.f, 0.f};

#pragma unroll 2
  for (int s = 0; s < NSTEP; ++s) {
    int t0 = 2 * s;
    int t1 = (t0 + 1 > 80) ? 80 : t0 + 1;        // pad tap: B is zero, A value irrelevant
    if (s + 1 < NSTEP) {
      __builtin_prefetch(fbase + (size_t)(s + 1) * 1024, 0, 3);
      __builtin_prefetch(hbase + tapOffB(2 * s + 2), 0, 3);
    }
    union { uint32_t u[8]; v16bf v; } A, B;
    const uint4 a0  = *(const uint4*)(hbase + tapOffB(t0));
    const uint4 a1  = *(const uint4*)(hbase + tapOffB(t1));
    const uint4 b0  = *(const uint4*)(fbase + (size_t)s * 1024);
    const uint4 b1v = *(const uint4*)(fbase + (size_t)s * 1024 + 16);
    A.u[0] = a0.x;  A.u[1] = a0.y;  A.u[2] = a0.z;  A.u[3] = a0.w;
    A.u[4] = a1.x;  A.u[5] = a1.y;  A.u[6] = a1.z;  A.u[7] = a1.w;
    B.u[0] = b0.x;  B.u[1] = b0.y;  B.u[2] = b0.z;  B.u[3] = b0.w;
    B.u[4] = b1v.x; B.u[5] = b1v.y; B.u[6] = b1v.z; B.u[7] = b1v.w;
    acc = __builtin_amdgcn_wmma_f32_16x16x32_bf16(
        false, A.v, false, B.v, (short)0, acc, false, false);
  }

  // epilogue: bias (b2.sum(0)) + ReLU, LDS transpose, coalesced stores
  float bias = b2[n] + b2[16 + n] + b2[32 + n];
  float* tb = xpose[wv];
#pragma unroll
  for (int v = 0; v < 8; ++v) {                   // D: m = v + 8*hl, n = lane%16
    float val = fmaxf(acc[v] + bias, 0.0f);
    tb[n * 17 + v + 8 * hl] = val;
  }
  __syncthreads();

  const int w0 = wt * 16;
  const int m0 = 8 * hl;
  float4 q0, q1;
  q0.x = tb[n * 17 + m0 + 0]; q0.y = tb[n * 17 + m0 + 1];
  q0.z = tb[n * 17 + m0 + 2]; q0.w = tb[n * 17 + m0 + 3];
  q1.x = tb[n * 17 + m0 + 4]; q1.y = tb[n * 17 + m0 + 5];
  q1.z = tb[n * 17 + m0 + 6]; q1.w = tb[n * 17 + m0 + 7];

  size_t ob = (size_t)n * OUT_NSTRIDE
            + ((size_t)((l * OD + d) * OH + hh)) * OW + w0 + m0;
  *(float4*)(out + ob) = q0;                      // m0..m0+3 always valid (>=4 cols left)
  if (OW - w0 - m0 >= 8)                          // skip only the wt==5, hl==1 tail
    *(float4*)(out + ob + 4) = q1;
}

// ---------------- host launcher ----------------
extern "C" void kernel_launch(void* const* d_in, const int* in_sizes, int n_in,
                              void* d_out, int out_size, void* d_ws, size_t ws_size,
                              hipStream_t stream) {
  (void)in_sizes; (void)n_in; (void)out_size; (void)ws_size;
  const float* x  = (const float*)d_in[0];
  const float* w1 = (const float*)d_in[1];
  const float* b1 = (const float*)d_in[2];
  const float* w2 = (const float*)d_in[3];
  const float* b2 = (const float*)d_in[4];
  float* out = (float*)d_out;

  // workspace: [ h bf16 (130 MB) | 4 KB guard (edge-tile over-reads) | w2 B-fragments (41 KB) ]
  unsigned short* h    = (unsigned short*)d_ws;
  unsigned short* frag = (unsigned short*)((char*)d_ws + HBYTES + 4096);

  pack_w2_kernel<<<(NSTEP * 32 * 16 + 255) / 256, 256, 0, stream>>>(w2, frag);
  conv1_kernel<<<(NP1 + 255) / 256, 256, 0, stream>>>(x, w1, b1, h);

  const int nwaves = OL * OD * OH * WT;           // 194,304 tiles, 4 waves/block
  conv2_kernel<<<nwaves / 4, 128, 0, stream>>>(h, frag, b2, out);
}
